// MultiHeadSelfAttention_85126251806804
// MI455X (gfx1250) — matrix operand order repaired and checked by reading, not compiled, not run on hardware
//
#include <hip/hip_runtime.h>

// ---------------------------------------------------------------------------
// MHA forward for gfx1250 (CDNA5): f16 WMMA (16x16x32) everywhere, flash-style
// streaming softmax (no LxL score materialization), fp32 accumulation.
// ---------------------------------------------------------------------------

typedef _Float16 h2   __attribute__((ext_vector_type(2)));
typedef _Float16 v8h  __attribute__((ext_vector_type(8)));
typedef _Float16 v16h __attribute__((ext_vector_type(16)));
typedef float    v8f  __attribute__((ext_vector_type(8)));

#define EMBED   1024
#define NHEADS  16
#define HDIM    64
#define BATCH   8
#define SEQ     1024
#define TOKENS  (BATCH * SEQ)      // 8192
#define QKVN    (3 * EMBED)        // 3072

static __device__ __forceinline__ v8f wmma_f16(v16h a, v16h b, v8f c) {
  // D = A(16x32 f16) * B(32x16 f16) + C(16x16 f32)
  return __builtin_amdgcn_wmma_f32_16x16x32_f16(false, a, false, b, (short)0, c,
                                                false, false);
}

// A-fragment (16x32, MxK) from row-major f16 matrix.
// lane = hf*16 + m;  VGPR j holds K = hf*16 + 2j (+0,+1).
// 16 contiguous f16 per lane -> compiler emits 2x b128 loads.
static __device__ __forceinline__ v16h load_fragA(const _Float16* __restrict__ base,
                                                  int ld, int lane) {
  const int m  = lane & 15;
  const int hf = lane >> 4;
  const _Float16* p = base + (size_t)m * ld;
  v16h a;
#pragma unroll
  for (int j = 0; j < 8; ++j) {
    const int k = (hf << 4) + (j << 1);
    h2 t = *(const h2*)(p + k);
    a[2 * j]     = t.x;
    a[2 * j + 1] = t.y;
  }
  return a;
}

// B-fragment (32x16, KxN) read from *transposed* storage BT[n][k] (row-major
// NxK). Contiguous k order, same as load_fragA: lane = hf*16 + n; VGPR j holds K = hf*16 + 2j (+0,+1).
// 16 contiguous f16 per lane -> 2x b128 loads.
static __device__ __forceinline__ v16h load_fragB_T(const _Float16* __restrict__ base,
                                                    int ld, int lane) {
  const int n  = lane & 15;
  const int hf = lane >> 4;
  const _Float16* p = base + (size_t)n * ld + (hf << 4);
  v16h b;
#pragma unroll
  for (int j = 0; j < 8; ++j) {
    h2 t = *(const h2*)(p + 2 * j);
    b[2 * j]     = t.x;
    b[2 * j + 1] = t.y;
  }
  return b;
}

// ---------------------------------------------------------------------------
// Precision-conversion kernels (fp32 -> f16), run once per launch.
// ---------------------------------------------------------------------------
__global__ void cvt_f32_to_f16(const float* __restrict__ in,
                               _Float16* __restrict__ out, int n) {
  int i = blockIdx.x * blockDim.x + threadIdx.x;
  if (i < n) out[i] = (_Float16)in[i];
}

// in[K][N] fp32 -> out[N][K] f16 (writes coalesced).
__global__ void cvt_transpose_f16(const float* __restrict__ in,
                                  _Float16* __restrict__ out, int K, int N) {
  int i = blockIdx.x * blockDim.x + threadIdx.x;
  if (i < K * N) {
    int n = i / K, k = i % K;
    out[i] = (_Float16)in[(size_t)k * N + n];
  }
}

// ---------------------------------------------------------------------------
// QKV projection: xh[8192][1024] @ wT(f16,[3072][1024]) + bias.
// Wave tile = 32(M) x 64(N): 2 A-frags x 4 B-frags -> 8 WMMAs per k-step,
// all fragments live simultaneously so loads clause together. Epilogue
// scatters into Qh[b][h][l][d] (pre-scaled 0.125), Kh[b][h][l][d],
// Vth[b][h][d][l] (V transposed for the attention PV matmul).
// ---------------------------------------------------------------------------
__global__ void gemm_qkv_kernel(const _Float16* __restrict__ xh,
                                const _Float16* __restrict__ wT,
                                const float* __restrict__ bias,
                                _Float16* __restrict__ Qh,
                                _Float16* __restrict__ Kh,
                                _Float16* __restrict__ Vth) {
  const int lane = threadIdx.x & 31;
  const int wave = blockIdx.x * (blockDim.x >> 5) + (threadIdx.x >> 5);
  const int NG = QKVN / 64;              // 48 column groups
  const int m0 = (wave / NG) * 32;
  const int n0 = (wave % NG) * 64;

  v8f acc[2][4] = {};
  for (int k0 = 0; k0 < EMBED; k0 += 32) {
    const v16h a0 = load_fragA(xh + (size_t)m0 * EMBED + k0, EMBED, lane);
    const v16h a1 = load_fragA(xh + (size_t)(m0 + 16) * EMBED + k0, EMBED, lane);
    const v16h b0 = load_fragB_T(wT + (size_t)(n0 +  0) * EMBED + k0, EMBED, lane);
    const v16h b1 = load_fragB_T(wT + (size_t)(n0 + 16) * EMBED + k0, EMBED, lane);
    const v16h b2 = load_fragB_T(wT + (size_t)(n0 + 32) * EMBED + k0, EMBED, lane);
    const v16h b3 = load_fragB_T(wT + (size_t)(n0 + 48) * EMBED + k0, EMBED, lane);
    acc[0][0] = wmma_f16(a0, b0, acc[0][0]);
    acc[0][1] = wmma_f16(a0, b1, acc[0][1]);
    acc[0][2] = wmma_f16(a0, b2, acc[0][2]);
    acc[0][3] = wmma_f16(a0, b3, acc[0][3]);
    acc[1][0] = wmma_f16(a1, b0, acc[1][0]);
    acc[1][1] = wmma_f16(a1, b1, acc[1][1]);
    acc[1][2] = wmma_f16(a1, b2, acc[1][2]);
    acc[1][3] = wmma_f16(a1, b3, acc[1][3]);
  }

  const int n15 = lane & 15;
  const int hf  = lane >> 4;
#pragma unroll
  for (int mt = 0; mt < 2; ++mt) {
    const int mb = m0 + mt * 16;
#pragma unroll
    for (int t = 0; t < 4; ++t) {
      const int f    = n0 + t * 16 + n15;   // global output feature (0..3071)
      const float bv = bias[f];
      const int part = f >> 10;             // 0=Q 1=K 2=V (uniform per wave+t)
      const int fh   = f & 1023;
      const int h    = fh >> 6;
      const int d    = fh & 63;
      if (part == 2) {
        // Vth[((b*H+h)*64+d)*SEQ + l]; rows r = 8 consecutive tokens -> B128
        const int tok0 = mb + (hf << 3);
        const int bb = tok0 >> 10, l0 = tok0 & 1023;
        v8h pk;
#pragma unroll
        for (int r = 0; r < 8; ++r) pk[r] = (_Float16)(acc[mt][t][r] + bv);
        *(v8h*)(Vth + (((size_t)bb * NHEADS + h) * HDIM + d) * SEQ + l0) = pk;
      } else {
        const float scl = (part == 0) ? 0.125f : 1.0f;  // fold 1/sqrt(64) into Q
        _Float16* dst = (part == 0) ? Qh : Kh;
#pragma unroll
        for (int r = 0; r < 8; ++r) {
          const int tok = mb + r + (hf << 3);
          const int bb = tok >> 10, l = tok & 1023;
          dst[(((size_t)bb * NHEADS + h) * SEQ + l) * HDIM + d] =
              (_Float16)((acc[mt][t][r] + bv) * scl);
        }
      }
    }
  }
}

// ---------------------------------------------------------------------------
// Flash attention. One wave owns (b, h, 32 queries = 2 query tiles), sharing
// each K/V fragment across both tiles (halves L2 traffic). We compute
// S^T = K·Q^T so the query index == lane&15: running max/sum are one scalar
// per lane per tile and the O^T rescale is lane-local. O^T = V^T·P^T.
// ---------------------------------------------------------------------------
__global__ void attn_flash_kernel(const _Float16* __restrict__ Qh,
                                  const _Float16* __restrict__ Kh,
                                  const _Float16* __restrict__ Vth,
                                  const unsigned char* __restrict__ mask,
                                  _Float16* __restrict__ ctxh) {
  const int lane = threadIdx.x & 31;
  const int wg   = blockIdx.x * (blockDim.x >> 5) + (threadIdx.x >> 5);
  const int b    = wg >> 9;            // / (H * L/32)
  const int rem  = wg & 511;
  const int h    = rem >> 5;
  const int q0   = (rem & 31) << 5;    // 32 queries per wave
  const int hf   = lane >> 4;

  const _Float16* Qb = Qh  + ((size_t)b * NHEADS + h) * SEQ * HDIM;
  const _Float16* Kb = Kh  + ((size_t)b * NHEADS + h) * SEQ * HDIM;
  const _Float16* Vb = Vth + ((size_t)b * NHEADS + h) * HDIM * SEQ;
  const unsigned char* mb = mask + (size_t)b * SEQ;

  // Q^T as B-fragments (per query tile: hd 0..31 / 32..63), live all loop.
  v16h qf[2][2];
#pragma unroll
  for (int qt = 0; qt < 2; ++qt) {
    qf[qt][0] = load_fragB_T(Qb + (size_t)(q0 + qt * 16) * HDIM +  0, HDIM, lane);
    qf[qt][1] = load_fragB_T(Qb + (size_t)(q0 + qt * 16) * HDIM + 32, HDIM, lane);
  }

  v8f o[2][4] = {};                    // O^T tiles: hd = t*16 + r + 8*hf
  float m_run[2] = {-1e30f, -1e30f};
  float l_run[2] = {0.0f, 0.0f};

  for (int key0 = 0; key0 < SEQ; key0 += 32) {
    // ---- K tile fragments (shared by both query tiles) ----
    const v16h ka0 = load_fragA(Kb + (size_t)key0 * HDIM +  0, HDIM, lane);
    const v16h ka1 = load_fragA(Kb + (size_t)key0 * HDIM + 32, HDIM, lane);
    const v16h kb0 = load_fragA(Kb + (size_t)(key0 + 16) * HDIM +  0, HDIM, lane);
    const v16h kb1 = load_fragA(Kb + (size_t)(key0 + 16) * HDIM + 32, HDIM, lane);

    // ---- S^T (keys x queries): 2 key-tiles x 2 query-tiles, hd over 2 WMMAs
    v8f sa[2] = {}, sb[2] = {};
#pragma unroll
    for (int qt = 0; qt < 2; ++qt) {
      sa[qt] = wmma_f16(ka0, qf[qt][0], sa[qt]);
      sa[qt] = wmma_f16(ka1, qf[qt][1], sa[qt]);
      sb[qt] = wmma_f16(kb0, qf[qt][0], sb[qt]);
      sb[qt] = wmma_f16(kb1, qf[qt][1], sb[qt]);
    }

    // ---- mask bytes (per key, shared across query tiles) ----
    unsigned char ma[8], mk[8];
#pragma unroll
    for (int r = 0; r < 8; ++r) {
      ma[r] = mb[key0 + r + (hf << 3)];
      mk[r] = mb[key0 + 16 + r + (hf << 3)];
    }

    // ---- streaming softmax per query tile; build P^T B-fragments ----
    v16h pf[2];
#pragma unroll
    for (int qt = 0; qt < 2; ++qt) {
      float pa[8], pb[8];
      float mx = -1e30f;
#pragma unroll
      for (int r = 0; r < 8; ++r) {
        pa[r] = ma[r] ? sa[qt][r] : -10000.0f;   // replace, as in reference
        pb[r] = mk[r] ? sb[qt][r] : -10000.0f;
        mx = fmaxf(mx, fmaxf(pa[r], pb[r]));
      }
      mx = fmaxf(mx, __shfl_xor(mx, 16, 32));
      const float m_new = fmaxf(m_run[qt], mx);
      const float alpha = __expf(m_run[qt] - m_new);
      float ssum = 0.0f;
#pragma unroll
      for (int r = 0; r < 8; ++r) {
        pa[r] = __expf(pa[r] - m_new);
        pb[r] = __expf(pb[r] - m_new);
        ssum += pa[r] + pb[r];
      }
      ssum += __shfl_xor(ssum, 16, 32);
      l_run[qt] = l_run[qt] * alpha + ssum;
      m_run[qt] = m_new;
#pragma unroll
      for (int t = 0; t < 4; ++t)
#pragma unroll
        for (int r = 0; r < 8; ++r) o[qt][t][r] *= alpha;

      // P^T C-layout -> B-fragment: exchange halves with shfl_xor(16).
      // lane<16 needs keys 0..15 (tile A), lane>=16 keys 16..31 (tile B).
#pragma unroll
      for (int r = 0; r < 8; ++r) {
        const float qa = __shfl_xor(pa[r], 16, 32);   // partner tile-A values
        const float qb = __shfl_xor(pb[r], 16, 32);   // partner tile-B values
        pf[qt][r]     = (_Float16)(hf ? qb : pa[r]);  // window key r
        pf[qt][8 + r] = (_Float16)(hf ? pb[r] : qa);  // window key r+8
      }
    }

    // ---- V^T fragments (shared), then O^T += V^T * P^T ----
    v16h vf[4];
#pragma unroll
    for (int t = 0; t < 4; ++t)
      vf[t] = load_fragA(Vb + (size_t)(t * 16) * SEQ + key0, SEQ, lane);
#pragma unroll
    for (int qt = 0; qt < 2; ++qt)
#pragma unroll
      for (int t = 0; t < 4; ++t)
        o[qt][t] = wmma_f16(vf[t], pf[qt], o[qt][t]);
  }

  // ---- finalize: O /= l, store ctx[b][l][h*64 + hd] as f16, 16B stores ----
#pragma unroll
  for (int qt = 0; qt < 2; ++qt) {
    const float inv_l = 1.0f / l_run[qt];
    const int tok = b * SEQ + q0 + qt * 16 + (lane & 15);
#pragma unroll
    for (int t = 0; t < 4; ++t) {
      v8h pk;
#pragma unroll
      for (int r = 0; r < 8; ++r) pk[r] = (_Float16)(o[qt][t][r] * inv_l);
      *(v8h*)(ctxh + (size_t)tok * EMBED + h * HDIM + t * 16 + (hf << 3)) = pk;
    }
  }
}

// ---------------------------------------------------------------------------
// Output projection: ctxh[8192][1024] @ w_outT(f16,[1024][1024]) + b_out -> f32
// Wave tile = 32(M) x 64(N), same structure as gemm_qkv.
// ---------------------------------------------------------------------------
__global__ void gemm_out_kernel(const _Float16* __restrict__ ctxh,
                                const _Float16* __restrict__ wT,
                                const float* __restrict__ bias,
                                float* __restrict__ out) {
  const int lane = threadIdx.x & 31;
  const int wave = blockIdx.x * (blockDim.x >> 5) + (threadIdx.x >> 5);
  const int NG = EMBED / 64;             // 16 column groups
  const int m0 = (wave / NG) * 32;
  const int n0 = (wave % NG) * 64;

  v8f acc[2][4] = {};
  for (int k0 = 0; k0 < EMBED; k0 += 32) {
    const v16h a0 = load_fragA(ctxh + (size_t)m0 * EMBED + k0, EMBED, lane);
    const v16h a1 = load_fragA(ctxh + (size_t)(m0 + 16) * EMBED + k0, EMBED, lane);
    const v16h b0 = load_fragB_T(wT + (size_t)(n0 +  0) * EMBED + k0, EMBED, lane);
    const v16h b1 = load_fragB_T(wT + (size_t)(n0 + 16) * EMBED + k0, EMBED, lane);
    const v16h b2 = load_fragB_T(wT + (size_t)(n0 + 32) * EMBED + k0, EMBED, lane);
    const v16h b3 = load_fragB_T(wT + (size_t)(n0 + 48) * EMBED + k0, EMBED, lane);
    acc[0][0] = wmma_f16(a0, b0, acc[0][0]);
    acc[0][1] = wmma_f16(a0, b1, acc[0][1]);
    acc[0][2] = wmma_f16(a0, b2, acc[0][2]);
    acc[0][3] = wmma_f16(a0, b3, acc[0][3]);
    acc[1][0] = wmma_f16(a1, b0, acc[1][0]);
    acc[1][1] = wmma_f16(a1, b1, acc[1][1]);
    acc[1][2] = wmma_f16(a1, b2, acc[1][2]);
    acc[1][3] = wmma_f16(a1, b3, acc[1][3]);
  }

  const int n15 = lane & 15;
  const int hf  = lane >> 4;
#pragma unroll
  for (int mt = 0; mt < 2; ++mt) {
#pragma unroll
    for (int t = 0; t < 4; ++t) {
      const int f    = n0 + t * 16 + n15;
      const float bv = bias[f];
#pragma unroll
      for (int r = 0; r < 8; ++r) {
        const int tok = m0 + mt * 16 + r + (hf << 3);
        out[(size_t)tok * EMBED + f] = acc[mt][t][r] + bv;
      }
    }
  }
}

// ---------------------------------------------------------------------------
// Host-side launch. Workspace carve-up (f16 elements, ~88 MB total):
//   xh 8M | wqkvT 3M | woutT 1M | Qh 8M | Kh 8M | Vth 8M | ctxh 8M
// ---------------------------------------------------------------------------
extern "C" void kernel_launch(void* const* d_in, const int* in_sizes, int n_in,
                              void* d_out, int out_size, void* d_ws, size_t ws_size,
                              hipStream_t stream) {
  const float* x      = (const float*)d_in[0];               // (8,1024,1024)
  const unsigned char* attn_mask = (const unsigned char*)d_in[1]; // (8,1024) bool
  const float* w_qkv  = (const float*)d_in[2];               // (1024,3072)
  const float* b_qkv  = (const float*)d_in[3];               // (3072,)
  const float* w_out  = (const float*)d_in[4];               // (1024,1024)
  const float* b_out  = (const float*)d_in[5];               // (1024,)
  float* out = (float*)d_out;

  _Float16* ws = (_Float16*)d_ws;
  const size_t N_TOK_D = (size_t)TOKENS * EMBED;             // 8388608
  _Float16* xh    = ws;                 ws += N_TOK_D;
  _Float16* wqkvT = ws;                 ws += (size_t)QKVN * EMBED;   // 3145728
  _Float16* woutT = ws;                 ws += (size_t)EMBED * EMBED;  // 1048576
  _Float16* Qh    = ws;                 ws += N_TOK_D;
  _Float16* Kh    = ws;                 ws += N_TOK_D;
  _Float16* Vth   = ws;                 ws += N_TOK_D;
  _Float16* ctxh  = ws;                 ws += N_TOK_D;

  const int T = 256;

  // 1) fp32 -> f16 conversions (weights transposed to NxK for B-fragments)
  cvt_f32_to_f16<<<(int)((N_TOK_D + T - 1) / T), T, 0, stream>>>(x, xh, (int)N_TOK_D);
  cvt_transpose_f16<<<(QKVN * EMBED + T - 1) / T, T, 0, stream>>>(w_qkv, wqkvT,
                                                                  EMBED, QKVN);
  cvt_transpose_f16<<<(EMBED * EMBED + T - 1) / T, T, 0, stream>>>(w_out, woutT,
                                                                   EMBED, EMBED);

  // 2) QKV projection (waves = 256 Mtiles * 48 Ngroups = 12288 -> 1536 blocks)
  gemm_qkv_kernel<<<(TOKENS / 32) * (QKVN / 64) / 8, T, 0, stream>>>(
      xh, wqkvT, b_qkv, Qh, Kh, Vth);

  // 3) Flash attention (waves = 8*16*32 = 4096 -> 512 blocks)
  attn_flash_kernel<<<BATCH * NHEADS * (SEQ / 32) / 8, T, 0, stream>>>(
      Qh, Kh, Vth, attn_mask, ctxh);

  // 4) Output projection (waves = 256 * 16 = 4096 -> 512 blocks)
  gemm_out_kernel<<<(TOKENS / 32) * (EMBED / 64) / 8, T, 0, stream>>>(
      ctxh, woutT, b_out, out);
}
